// AttentionLayer_25967372271679
// MI455X (gfx1250) — matrix operand order repaired and checked by reading, not compiled, not run on hardware
//
#include <hip/hip_runtime.h>
#include <hip/hip_bf16.h>

typedef __attribute__((ext_vector_type(16))) _Float16 v16h;
typedef __attribute__((ext_vector_type(8)))  float    v8f;
typedef int v4i_ __attribute__((vector_size(16)));
typedef __attribute__((address_space(1))) v4i_* as1_v4i;
typedef __attribute__((address_space(3))) v4i_* as3_v4i;

#define B_   16
#define T_   12
#define N_   307
#define D_   256
#define H_   8
#define HD_  32
#define NP_  320                 // N padded to 20 * 16
#define BT_  (B_ * T_)           // 192
#define MP_  (BT_ * NP_)         // 61440 padded rows
#define SZQ  (BT_ * H_ * NP_ * HD_)   // 15,728,640 f16 elems per tensor
#define SCALE 0.17677669529663687f    // 1/sqrt(32)

// 16B-aligned LDS strides (in halves) so fragments load as ds_load_b128
#define XS_STRIDE 40    // 80B rows: bank step 20 -> conflict-free b128
#define WS_STRIDE 264   // 528B rows: bank step 4 -> worst 2-way on b128
#define SP_STRIDE 328   // 656B rows: bank step 36 -> conflict-free b128

// ---------------------------------------------------------------------------
// Async global->LDS (CDNA5). The builtin takes v4i pointers in AS1/AS3.
// ---------------------------------------------------------------------------
#if __has_builtin(__builtin_amdgcn_global_load_async_to_lds_b128) && \
    __has_builtin(__builtin_amdgcn_s_wait_asynccnt)
#define HAVE_ASYNC 1
#else
#define HAVE_ASYNC 0
#endif

__device__ __forceinline__ void copy16_async(const _Float16* g, _Float16* l) {
#if HAVE_ASYNC
    __builtin_amdgcn_global_load_async_to_lds_b128((as1_v4i)g, (as3_v4i)l, 0, 0);
#else
    *(uint4*)l = *(const uint4*)g;
#endif
}

__device__ __forceinline__ void async_wait_all() {
#if HAVE_ASYNC
    __builtin_amdgcn_s_wait_asynccnt(0);
#endif
}

// ---------------------------------------------------------------------------
// Kernel 0: f32 -> f16 weight conversion (one-shot, 4 x 64K elements)
// ---------------------------------------------------------------------------
__global__ __launch_bounds__(256)
void cvt_f32_f16_kernel(const float* __restrict__ src,
                        _Float16* __restrict__ dst, int n) {
    int i = (blockIdx.x * 256 + threadIdx.x) * 4;
    if (i + 3 < n) {
        float4 v = *(const float4*)(src + i);
        dst[i + 0] = (_Float16)v.x;
        dst[i + 1] = (_Float16)v.y;
        dst[i + 2] = (_Float16)v.z;
        dst[i + 3] = (_Float16)v.w;
    }
}

// ---------------------------------------------------------------------------
// QKV projection: block = 256 thr (8 waves), tile = 128 x 256, K-slices of 32.
// A (activations) staged with f32->f16 VALU converts; B (weights, already
// f16 in ws) staged with async global->LDS B128 DMA.
// ---------------------------------------------------------------------------
__global__ __launch_bounds__(256)
void qkv_proj_kernel(const float* __restrict__ X,      // [B*T*N, 256] f32
                     const _Float16* __restrict__ Wh,  // [256, 256] f16
                     const float* __restrict__ bias,   // [256]
                     _Float16* __restrict__ dst,
                     int transposed)  // 0: [bt][h][np][hd], 1: [bt][h][hd][np]
{
    __shared__ _Float16 Xs[128][XS_STRIDE];
    __shared__ _Float16 Ws[32][WS_STRIDE];

    const int tid  = threadIdx.x;
    const int wave = tid >> 5;
    const int lane = tid & 31;
    const int hi   = lane >> 4;
    const int lr   = lane & 15;
    const int rowBase = blockIdx.x * 128;

    // per-thread staging coords
    const int xr  = tid >> 1;             // X: 2 thr/row, 16 f32 each
    const int xc0 = (tid & 1) * 16;
    const int xrp = rowBase + xr;
    const int xbt = xrp / NP_, xnp = xrp % NP_;
    const float* xsrc = (xnp < N_)
        ? X + ((size_t)(xbt * N_ + xnp)) * D_ + xc0 : (const float*)0;

    v8f zero = {};
    v8f acc[16];
#pragma unroll
    for (int i = 0; i < 16; ++i) acc[i] = zero;

    for (int kb = 0; kb < 8; ++kb) {
        // B tile: async DMA 32x256 f16 (4 x 16B chunks per thread)
#pragma unroll
        for (int c = 0; c < 4; ++c) {
            int chunk = tid + c * 256;            // 1024 chunks of 8 halves
            int r  = chunk >> 5;
            int c8 = (chunk & 31) * 8;
            copy16_async(Wh + (size_t)(kb * 32 + r) * D_ + c8, &Ws[r][c8]);
        }
        // A tile: f32 load + convert (padded rows are zero)
        if (xsrc) {
            const float* s = xsrc + kb * 32;
            if (kb < 7) __builtin_prefetch(s + 32, 0, 0);   // next K-slice
#pragma unroll
            for (int j = 0; j < 16; j += 2) {
                float2 v = *(const float2*)(s + j);
                Xs[xr][xc0 + j]     = (_Float16)v.x;
                Xs[xr][xc0 + j + 1] = (_Float16)v.y;
            }
        } else {
#pragma unroll
            for (int j = 0; j < 16; ++j) Xs[xr][xc0 + j] = (_Float16)0.0f;
        }
        if (kb < 7) __builtin_prefetch(Wh + (size_t)(kb * 32 + 32) * D_ + tid * 8, 0, 0);
        async_wait_all();
        __syncthreads();

        // A fragment (ISA layout): two 16B-aligned b128 groups per lane
        v16h a;
#pragma unroll
        for (int i = 0; i < 16; ++i) {
            int k = (i < 8 ? i : i + 8) + hi * 8;
            a[i] = Xs[wave * 16 + lr][k];
        }
#pragma unroll
        for (int ct = 0; ct < 16; ++ct) {
            v16h b;
#pragma unroll
            for (int i = 0; i < 16; ++i) b[i] = Ws[(i < 8 ? i : i + 8) + hi * 8][ct * 16 + lr];
            acc[ct] = __builtin_amdgcn_wmma_f32_16x16x32_f16(
                false, a, false, b, (short)0, acc[ct], false, false);
        }
        __syncthreads();
    }

    // epilogue: bias + f16 store into head-split (optionally transposed) layout
#pragma unroll
    for (int ct = 0; ct < 16; ++ct) {
        int col = ct * 16 + lr;
        float bv = bias[col];
        int h = col >> 5, hd = col & 31;
#pragma unroll
        for (int r = 0; r < 8; ++r) {
            int rp = rowBase + wave * 16 + r + hi * 8;
            int bt = rp / NP_, np = rp % NP_;
            float v = acc[ct][r] + bv;
            size_t off;
            if (transposed) off = ((size_t)((bt * H_ + h) * HD_ + hd)) * NP_ + np;
            else            off = ((size_t)((bt * H_ + h) * NP_ + np)) * HD_ + hd;
            dst[off] = (_Float16)v;
        }
    }
}

// ---------------------------------------------------------------------------
// Fused attention: one block (4 waves) per (bt, h, 16-query tile).
// ---------------------------------------------------------------------------
__global__ __launch_bounds__(128)
void attn_kernel(const _Float16* __restrict__ qh,  // [bt][h][np][hd]
                 const _Float16* __restrict__ kt,  // [bt][h][hd][np]
                 const _Float16* __restrict__ vh,  // [bt][h][np][hd]
                 const unsigned char* __restrict__ geo,
                 const unsigned char* __restrict__ sem,
                 _Float16* __restrict__ ctx)       // [bt*NP][256]
{
    __shared__ _Float16 qs[16][XS_STRIDE];
    __shared__ float    sS[16][NP_ + 8];
    __shared__ _Float16 sP[16][SP_STRIDE];
    __shared__ float    red[16][8];
    __shared__ float    rowsum[16];
    __shared__ float    po[4][16][32];

    const int tid  = threadIdx.x;
    const int wave = tid >> 5;
    const int lane = tid & 31;
    const int hi   = lane >> 4;
    const int lr   = lane & 15;

    int bx = blockIdx.x;
    const int qt = bx % 20;  bx /= 20;
    const int h  = bx % 8;
    const int bt = bx / 8;

    // stage the 16x32 q tile via async DMA (64 chunks of 16B)
    if (tid < 64) {
        int r = tid >> 2, c8 = (tid & 3) * 8;
        copy16_async(qh + ((size_t)((bt * H_ + h) * NP_) + qt * 16 + r) * HD_ + c8,
                     &qs[r][c8]);
    }
    async_wait_all();
    __syncthreads();

    v16h a;
#pragma unroll
    for (int i = 0; i < 16; ++i) {
        int k = (i < 8 ? i : i + 8) + hi * 8;
        a[i] = qs[lr][k];
    }

    // scores: 20 chunks of 16 keys; one WMMA per chunk (K = HD = 32)
    const _Float16* ktb = kt + (size_t)((bt * H_ + h) * HD_) * NP_;
    for (int c = wave; c < 20; c += 4) {
        v16h b;
#pragma unroll
        for (int i = 0; i < 16; ++i)
            b[i] = ktb[(size_t)((i < 8 ? i : i + 8) + hi * 8) * NP_ + c * 16 + lr];
        v8f s = {};
        s = __builtin_amdgcn_wmma_f32_16x16x32_f16(
            false, a, false, b, (short)0, s, false, false);
#pragma unroll
        for (int r = 0; r < 8; ++r)
            sS[r + hi * 8][c * 16 + lr] = s[r] * SCALE;
    }
    __syncthreads();

    // masked softmax (unnormalized): 8 threads per row, 40 cols each
    {
        int row = tid >> 3, sub = tid & 7;
        int qn = qt * 16 + row; if (qn >= N_) qn = N_ - 1;
        const unsigned char* gm = geo + (size_t)qn * N_;
        const unsigned char* sm = sem + (size_t)qn * N_;
        int c0 = sub * 40;
        float vals[40];
        float mx = -1e30f;
#pragma unroll 4
        for (int j = 0; j < 40; ++j) {
            int kn = c0 + j;
            float v = -1e30f;
            if (kn < N_ && !gm[kn] && !sm[kn]) v = sS[row][kn];
            vals[j] = v;
            mx = fmaxf(mx, v);
        }
        red[row][sub] = mx;
        __syncthreads();
        float rmx = red[row][0];
#pragma unroll
        for (int j = 1; j < 8; ++j) rmx = fmaxf(rmx, red[row][j]);
        float sum = 0.f;
#pragma unroll 4
        for (int j = 0; j < 40; ++j) {
            float e = (vals[j] > -1e29f) ? __expf(vals[j] - rmx) : 0.f;
            sum += e;
            sP[row][c0 + j] = (_Float16)e;
        }
        __syncthreads();
        red[row][sub] = sum;
        __syncthreads();
        if (sub == 0) {
            float s2 = 0.f;
#pragma unroll
            for (int j = 0; j < 8; ++j) s2 += red[row][j];
            rowsum[row] = s2;
        }
    }
    __syncthreads();

    // P @ V: 10 chunks of 32 keys, 2 WMMAs per chunk (HD split 16+16)
    v8f o0 = {}, o1 = {};
    for (int c2 = wave; c2 < 10; c2 += 4) {
        v16h pa;
#pragma unroll
        for (int i = 0; i < 16; ++i) {
            int k = (i < 8 ? i : i + 8) + hi * 8;
            pa[i] = sP[lr][c2 * 32 + k];
        }
        const _Float16* vb =
            vh + ((size_t)((bt * H_ + h) * NP_) + c2 * 32) * HD_;
        v16h b0, b1;
#pragma unroll
        for (int i = 0; i < 16; ++i) {
            int k = (i < 8 ? i : i + 8) + hi * 8;
            b0[i] = vb[(size_t)k * HD_ + lr];
            b1[i] = vb[(size_t)k * HD_ + 16 + lr];
        }
        o0 = __builtin_amdgcn_wmma_f32_16x16x32_f16(
            false, pa, false, b0, (short)0, o0, false, false);
        o1 = __builtin_amdgcn_wmma_f32_16x16x32_f16(
            false, pa, false, b1, (short)0, o1, false, false);
    }
#pragma unroll
    for (int r = 0; r < 8; ++r) {
        po[wave][r + hi * 8][lr]      = o0[r];
        po[wave][r + hi * 8][16 + lr] = o1[r];
    }
    __syncthreads();

    // reduce 4 wave partials, apply 1/rowsum, write f16 context
    {
        int row = tid >> 3;
        int c0  = (tid & 7) * 4;
        float inv = 1.0f / rowsum[row];
        _Float16* dst =
            ctx + ((size_t)(bt * NP_ + qt * 16 + row)) * D_ + h * HD_ + c0;
#pragma unroll
        for (int j = 0; j < 4; ++j) {
            float s = po[0][row][c0 + j] + po[1][row][c0 + j] +
                      po[2][row][c0 + j] + po[3][row][c0 + j];
            dst[j] = (_Float16)(s * inv);
        }
    }
}

// ---------------------------------------------------------------------------
// Output projection: ctx (f16) @ Wo(f16) + bo -> f32. Both tiles async-DMA'd.
// ---------------------------------------------------------------------------
__global__ __launch_bounds__(256)
void out_proj_kernel(const _Float16* __restrict__ ctx, // [MP, 256] f16
                     const _Float16* __restrict__ Woh, // [256,256] f16
                     const float* __restrict__ bo,
                     float* __restrict__ out)          // [B*T*N, 256] f32
{
    __shared__ _Float16 Xs[128][XS_STRIDE];
    __shared__ _Float16 Ws[32][WS_STRIDE];

    const int tid  = threadIdx.x;
    const int wave = tid >> 5;
    const int lane = tid & 31;
    const int hi   = lane >> 4;
    const int lr   = lane & 15;
    const int rowBase = blockIdx.x * 128;

    v8f zero = {};
    v8f acc[16];
#pragma unroll
    for (int i = 0; i < 16; ++i) acc[i] = zero;

    for (int kb = 0; kb < 8; ++kb) {
        // A tile: 128x32 f16 = 512 chunks of 16B, 2 per thread
#pragma unroll
        for (int c = 0; c < 2; ++c) {
            int chunk = tid + c * 256;
            int r  = chunk >> 2;
            int c8 = (chunk & 3) * 8;
            copy16_async(ctx + (size_t)(rowBase + r) * D_ + kb * 32 + c8,
                         &Xs[r][c8]);
        }
        // B tile: 32x256 f16 = 1024 chunks of 16B, 4 per thread
#pragma unroll
        for (int c = 0; c < 4; ++c) {
            int chunk = tid + c * 256;
            int r  = chunk >> 5;
            int c8 = (chunk & 31) * 8;
            copy16_async(Woh + (size_t)(kb * 32 + r) * D_ + c8, &Ws[r][c8]);
        }
        if (kb < 7) {
            __builtin_prefetch(ctx + (size_t)(rowBase + (tid >> 1)) * D_ + kb * 32 + 32, 0, 0);
            __builtin_prefetch(Woh + (size_t)(kb * 32 + 32) * D_ + tid * 8, 0, 0);
        }
        async_wait_all();
        __syncthreads();

        v16h a;
#pragma unroll
        for (int i = 0; i < 16; ++i) {
            int k = (i < 8 ? i : i + 8) + hi * 8;
            a[i] = Xs[wave * 16 + lr][k];
        }
#pragma unroll
        for (int ct = 0; ct < 16; ++ct) {
            v16h b;
#pragma unroll
            for (int i = 0; i < 16; ++i) b[i] = Ws[(i < 8 ? i : i + 8) + hi * 8][ct * 16 + lr];
            acc[ct] = __builtin_amdgcn_wmma_f32_16x16x32_f16(
                false, a, false, b, (short)0, acc[ct], false, false);
        }
        __syncthreads();
    }

#pragma unroll
    for (int ct = 0; ct < 16; ++ct) {
        int col = ct * 16 + lr;
        float bv = bo[col];
#pragma unroll
        for (int r = 0; r < 8; ++r) {
            int rp = rowBase + wave * 16 + r + hi * 8;
            int bt = rp / NP_, np = rp % NP_;
            if (np < N_) {
                out[((size_t)(bt * N_ + np)) * D_ + col] = acc[ct][r] + bv;
            }
        }
    }
}

// ---------------------------------------------------------------------------

extern "C" void kernel_launch(void* const* d_in, const int* in_sizes, int n_in,
                              void* d_out, int out_size, void* d_ws, size_t ws_size,
                              hipStream_t stream) {
    (void)in_sizes; (void)n_in; (void)out_size; (void)ws_size;

    const float* query = (const float*)d_in[0];
    const float* key   = (const float*)d_in[1];
    const float* value = (const float*)d_in[2];
    const unsigned char* geo = (const unsigned char*)d_in[3];
    const unsigned char* sem = (const unsigned char*)d_in[4];
    const float* Wq = (const float*)d_in[5];
    const float* bq = (const float*)d_in[6];
    const float* Wk = (const float*)d_in[7];
    const float* bk = (const float*)d_in[8];
    const float* Wv = (const float*)d_in[9];
    const float* bv = (const float*)d_in[10];
    const float* Wo = (const float*)d_in[11];
    const float* bo = (const float*)d_in[12];
    float* out = (float*)d_out;

    _Float16* qh  = (_Float16*)d_ws;       // [bt][h][np][hd]
    _Float16* kt  = qh + SZQ;              // [bt][h][hd][np]
    _Float16* vh  = kt + SZQ;              // [bt][h][np][hd]
    _Float16* ctx = vh + SZQ;              // [bt*NP][256]
    _Float16* wqh = ctx + SZQ;             // 4 x [256,256] f16 weights
    _Float16* wkh = wqh + D_ * D_;
    _Float16* wvh = wkh + D_ * D_;
    _Float16* woh = wvh + D_ * D_;

    dim3 blk(256);
    dim3 grd(MP_ / 128);   // 480 workgroups
    dim3 cgrd(D_ * D_ / 1024);

    hipLaunchKernelGGL(cvt_f32_f16_kernel, cgrd, blk, 0, stream, Wq, wqh, D_ * D_);
    hipLaunchKernelGGL(cvt_f32_f16_kernel, cgrd, blk, 0, stream, Wk, wkh, D_ * D_);
    hipLaunchKernelGGL(cvt_f32_f16_kernel, cgrd, blk, 0, stream, Wv, wvh, D_ * D_);
    hipLaunchKernelGGL(cvt_f32_f16_kernel, cgrd, blk, 0, stream, Wo, woh, D_ * D_);

    hipLaunchKernelGGL(qkv_proj_kernel, grd, blk, 0, stream, query, wqh, bq, qh, 0);
    hipLaunchKernelGGL(qkv_proj_kernel, grd, blk, 0, stream, key,   wkh, bk, kt, 1);
    hipLaunchKernelGGL(qkv_proj_kernel, grd, blk, 0, stream, value, wvh, bv, vh, 0);

    hipLaunchKernelGGL(attn_kernel, dim3(BT_ * H_ * 20), dim3(128), 0, stream,
                       qh, kt, vh, geo, sem, ctx);

    hipLaunchKernelGGL(out_proj_kernel, grd, blk, 0, stream, ctx, woh, bo, out);
}
